// _SelfAttn_46420006536004
// MI455X (gfx1250) — matrix operand order repaired and checked by reading, not compiled, not run on hardware
//
#include <hip/hip_runtime.h>

#define BATCH  8
#define SEQ    2048
#define DMODEL 512
#define BSROWS (BATCH * SEQ)                 // 16384
#define SCALEF 0.044194173824159216f         // 1/sqrt(512)

typedef __attribute__((ext_vector_type(16))) __bf16        v16bf;
typedef __attribute__((ext_vector_type(8)))  float         v8f;
typedef __attribute__((ext_vector_type(8)))  unsigned int  v8u;
typedef __attribute__((ext_vector_type(4)))  unsigned int  u32x4;
typedef __attribute__((ext_vector_type(8)))  int           i32x8;
typedef __attribute__((ext_vector_type(4)))  int           i32x4;

// Packed fp32x2 -> bf16x2 (RNE) in one VALU op.
__device__ __forceinline__ unsigned int pack2bf(float a, float b) {
  unsigned int r;
  asm("v_cvt_pk_bf16_f32 %0, %1, %2" : "=v"(r) : "v"(a), "v"(b));
  return r;
}
__device__ __forceinline__ unsigned short f2bf(float f) {
  return (unsigned short)pack2bf(f, 0.0f);
}
__device__ __forceinline__ v16bf mkfrag(uint4 a, uint4 b) {
  v8u u;
  u[0] = a.x; u[1] = a.y; u[2] = a.z; u[3] = a.w;
  u[4] = b.x; u[5] = b.y; u[6] = b.z; u[7] = b.w;
  return __builtin_bit_cast(v16bf, u);
}
__device__ __forceinline__ v8f wmma_bf16(v16bf a, v16bf b, v8f c) {
  return __builtin_amdgcn_wmma_f32_16x16x32_bf16(
      false, a, false, b, (short)0, c, false, false);
}

// ---------------------------------------------------------------------------
// Kernel 1: projections.  Y = x @ W^T for W in {Wq, Wk, Wv} (blockIdx.z).
// Q, K stored bf16 [BS, D]; V stored transposed bf16 [B, D, S].
// ---------------------------------------------------------------------------
__global__ __launch_bounds__(256) void proj_kernel(
    const float* __restrict__ x,  const float* __restrict__ Wq,
    const float* __restrict__ Wk, const float* __restrict__ Wv,
    unsigned short* __restrict__ Qb, unsigned short* __restrict__ Kb,
    unsigned short* __restrict__ Vt)
{
  __shared__ unsigned int Xs[128][16];   // 128 rows x 32 bf16 (packed pairs)
  __shared__ unsigned int Ws[64][16];    //  64 rows x 32 bf16

  const int t    = threadIdx.x;
  const int wave = t >> 5, lane = t & 31;
  const int hi   = lane >> 4, l15 = lane & 15;
  const int which = blockIdx.z;
  const float* W = (which == 0) ? Wq : ((which == 1) ? Wk : Wv);
  const int row0 = blockIdx.x * 128;
  const int n0   = blockIdx.y * 64;

  const v8f vz = {0.f,0.f,0.f,0.f,0.f,0.f,0.f,0.f};
  v8f acc[4] = {vz, vz, vz, vz};

  const int xr = t >> 1, xc = (t & 1) * 16;   // x-tile loader coords
  const int wr = t >> 2, wc = (t & 3) * 8;    // W-tile loader coords

  for (int ks = 0; ks < DMODEL; ks += 32) {
    __syncthreads();
    { // x tile: fp32 -> bf16 into LDS
      const float* src = x + (size_t)(row0 + xr) * DMODEL + ks + xc;
      #pragma unroll
      for (int j = 0; j < 4; ++j) {
        float4 f = *(const float4*)(src + 4 * j);
        Xs[xr][(xc >> 1) + 2*j + 0] = pack2bf(f.x, f.y);
        Xs[xr][(xc >> 1) + 2*j + 1] = pack2bf(f.z, f.w);
      }
    }
    { // W tile (row-major [e][d] -> B pairs contiguous)
      const float* src = W + (size_t)(n0 + wr) * DMODEL + ks + wc;
      #pragma unroll
      for (int j = 0; j < 2; ++j) {
        float4 f = *(const float4*)(src + 4 * j);
        Ws[wr][(wc >> 1) + 2*j + 0] = pack2bf(f.x, f.y);
        Ws[wr][(wc >> 1) + 2*j + 1] = pack2bf(f.z, f.w);
      }
    }
    __syncthreads();
    // Batch fragment loads, then run WMMAs back-to-back.
    const int arow = wave * 16 + l15;
    uint4 a0 = *(const uint4*)&Xs[arow][4 * hi];
    uint4 a1 = *(const uint4*)&Xs[arow][8 + 4 * hi];
    uint4 bf0[4], bf1[4];
    #pragma unroll
    for (int nt = 0; nt < 4; ++nt) {
      const int brow = nt * 16 + l15;
      bf0[nt] = *(const uint4*)&Ws[brow][4 * hi];
      bf1[nt] = *(const uint4*)&Ws[brow][8 + 4 * hi];
    }
    v16bf A = mkfrag(a0, a1);
    #pragma unroll
    for (int nt = 0; nt < 4; ++nt)
      acc[nt] = wmma_bf16(A, mkfrag(bf0[nt], bf1[nt]), acc[nt]);
  }

  // Epilogue: C/D layout (m = r + 8*hi, n = l15).  One base pointer per case;
  // all per-(nt,r) offsets are compile-time immediates.
  if (which == 2) {
    const int bb = row0 >> 11;                            // batch of this block
    const int s0 = (row0 & (SEQ - 1)) + wave * 16 + 8 * hi;
    unsigned short* base = Vt + ((size_t)bb * DMODEL + n0 + l15) * SEQ + s0;
    #pragma unroll
    for (int nt = 0; nt < 4; ++nt)
      #pragma unroll
      for (int r = 0; r < 8; ++r)
        base[(size_t)(nt * 16) * SEQ + r] = f2bf(acc[nt][r]);
  } else {
    unsigned short* dst = (which == 0) ? Qb : Kb;
    unsigned short* base =
        dst + (size_t)(row0 + wave * 16 + 8 * hi) * DMODEL + n0 + l15;
    #pragma unroll
    for (int nt = 0; nt < 4; ++nt)
      #pragma unroll
      for (int r = 0; r < 8; ++r)
        base[r * DMODEL + nt * 16] = f2bf(acc[nt][r]);
  }
}

// ---------------------------------------------------------------------------
// Kernel 2: flash attention.  Block = 32 queries, 8 waves = 2(q) x 4(d-slice).
// K chunks staged into LDS by the Tensor Data Mover (TENSORcnt tracked).
// ---------------------------------------------------------------------------
__global__ __launch_bounds__(256) void attn_kernel(
    const unsigned short* __restrict__ Qb, const unsigned short* __restrict__ Kb,
    const unsigned short* __restrict__ Vt, const float* __restrict__ bias,
    const unsigned char* __restrict__ kpm, float* __restrict__ out)
{
  __shared__ unsigned int Kt[32][256];     // 32 keys x 512 bf16  (32 KB)
  __shared__ float        Ssc[2][16][32];  // reduced score tiles  (4 KB)
  __shared__ unsigned int Ps[2][16][16];   // P tiles, 16x32 bf16  (2 KB)
  __shared__ float        mstat[2][16], lstat[2][16], alph[2][16];
  __shared__ float        smask[32];

  const int t    = threadIdx.x;
  const int wave = t >> 5, lane = t & 31;
  const int hi   = lane >> 4, l15 = lane & 15;
  const int qw   = wave >> 2, dw = wave & 3;
  const int b    = blockIdx.y;
  const int q0   = blockIdx.x * 32;
  const float NEG_INF = -__builtin_inff();
  const float bv = bias[0];

  // Preload Q A-fragments for this wave's 16 q-rows x 128 d-slice (32 VGPRs)
  uint4 qf0[4], qf1[4];
  {
    const unsigned short* qrow =
        Qb + ((size_t)(b * SEQ) + q0 + qw * 16 + l15) * DMODEL + dw * 128;
    #pragma unroll
    for (int ks = 0; ks < 4; ++ks) {
      qf0[ks] = *(const uint4*)(const void*)(qrow + ks * 32 + 8 * hi);
      qf1[ks] = *(const uint4*)(const void*)(qrow + ks * 32 + 16 + 8 * hi);
    }
  }
  // Loop-invariant V base: per-nt offsets become immediates (nt*16*SEQ elems).
  const unsigned short* vbase =
      Vt + ((size_t)b * DMODEL + dw * 128 + l15) * SEQ + 16 * hi;

  const v8f vz = {0.f,0.f,0.f,0.f,0.f,0.f,0.f,0.f};
  v8f acc[8] = {vz, vz, vz, vz, vz, vz, vz, vz};  // 16 q x 128 d accumulator

  if (t < 32) { (&mstat[0][0])[t] = NEG_INF; (&lstat[0][0])[t] = 0.f; }

  // Invariant parts of the TDM descriptor (1-D tile: 32 keys x 512 bf16).
  const unsigned int kt_lds = (unsigned int)(size_t)(&Kt[0][0]);
  const i32x4 zg4 = {0, 0, 0, 0};
  const i32x8 zg8 = {0, 0, 0, 0, 0, 0, 0, 0};
  i32x8 g1;
  g1[0] = 0x00010000;            // workgroup_mask=0, data_size=1 (2 bytes)
  g1[1] = (int)(16384u << 16);   // tensor_dim0[15:0] = 16384 (low half)
  g1[2] = (int)(1u << 16);       // tensor_dim0 hi = 0, tensor_dim1 = 1 (low)
  g1[3] = (int)(16384u << 16);   // tensor_dim1 hi = 0, tile_dim0 = 16384
  g1[4] = 0;                     // tile_dim1 = tile_dim2 = 0 (unused)
  g1[5] = 16384;                 // tensor_dim0_stride (low 32)
  g1[6] = 0;
  g1[7] = 0;

  for (int kc = 0; kc < SEQ; kc += 32) {
    __syncthreads();             // all waves done reading Kt from prev chunk
    if (wave == 0) {             // one TDM DMA stages the 32 KB K chunk
      unsigned long long ga =
          (unsigned long long)(size_t)(Kb + ((size_t)(b * SEQ) + kc) * DMODEL);
      u32x4 g0;
      g0[0] = 1u;                                      // count=1
      g0[1] = kt_lds;                                  // lds_addr
      g0[2] = (unsigned int)ga;                        // global_addr lo
      g0[3] = (unsigned int)(ga >> 32) | 0x80000000u;  // addr hi | type=2
      __builtin_amdgcn_tensor_load_to_lds(g0, g1, zg4, zg4, zg8, 0);
      __builtin_amdgcn_s_wait_tensorcnt(0);
    }
    if (t < 32) smask[t] = kpm[(size_t)b * SEQ + kc + t] ? NEG_INF : 0.f;
    { // zero score reduction buffer
      float* z = &Ssc[0][0][0];
      #pragma unroll
      for (int i = 0; i < 4; ++i) z[t + 256 * i] = 0.f;
    }
    __syncthreads();

    // --- partial scores over this wave's 128-d slice, reduced via ds_add ---
    #pragma unroll
    for (int nt = 0; nt < 2; ++nt) {
      const unsigned int* krow = &Kt[nt * 16 + l15][0];
      uint4 kb0[4], kb1[4];
      #pragma unroll
      for (int ks = 0; ks < 4; ++ks) {   // batch loads, single dscnt wait
        kb0[ks] = *(const uint4*)&krow[dw * 64 + ks * 16 + 4 * hi];
        kb1[ks] = *(const uint4*)&krow[dw * 64 + ks * 16 + 8 + 4 * hi];
      }
      v8f sacc = vz;
      #pragma unroll
      for (int ks = 0; ks < 4; ++ks)
        sacc = wmma_bf16(mkfrag(qf0[ks], qf1[ks]), mkfrag(kb0[ks], kb1[ks]), sacc);
      #pragma unroll
      for (int r = 0; r < 8; ++r)
        atomicAdd(&Ssc[qw][r + 8 * hi][nt * 16 + l15], sacc[r]);
    }
    __syncthreads();

    // --- online softmax: one lane per query row ---
    if (dw == 0 && lane < 16) {
      const int m = lane;
      float mold = mstat[qw][m];
      float cmax = NEG_INF;
      #pragma unroll 8
      for (int k = 0; k < 32; ++k) {
        float s = Ssc[qw][m][k] * SCALEF + bv + smask[k];
        Ssc[qw][m][k] = s;
        cmax = fmaxf(cmax, s);
      }
      float mnew = fmaxf(mold, cmax);
      float alpha = 1.0f;
      if (mnew == NEG_INF) {           // every key so far masked: P row = 0
        #pragma unroll
        for (int k = 0; k < 16; ++k) Ps[qw][m][k] = 0u;
      } else {
        alpha = __expf(mold - mnew);   // mold == -inf -> 0
        float rs = 0.f;
        #pragma unroll 8
        for (int k = 0; k < 32; k += 2) {
          float p0 = __expf(Ssc[qw][m][k]     - mnew);  // masked -> exp(-inf)=0
          float p1 = __expf(Ssc[qw][m][k + 1] - mnew);
          rs += p0 + p1;
          Ps[qw][m][k >> 1] = pack2bf(p0, p1);
        }
        lstat[qw][m] = lstat[qw][m] * alpha + rs;
        mstat[qw][m] = mnew;
      }
      alph[qw][m] = alpha;
    }
    __syncthreads();

    // --- rescale accumulator and accumulate P x V ---
    float av[8];
    #pragma unroll
    for (int r = 0; r < 8; ++r) av[r] = alph[qw][r + 8 * hi];
    const unsigned int* prow = &Ps[qw][l15][0];
    uint4 p0 = *(const uint4*)&prow[8 * hi];
    uint4 p1 = *(const uint4*)&prow[8 * hi + 4];
    v16bf Pa = mkfrag(p0, p1);
    #pragma unroll
    for (int nt = 0; nt < 8; ++nt) {
      const unsigned short* vrow = vbase + kc + (size_t)(nt * 16) * SEQ;
      uint4 v0 = *(const uint4*)(const void*)(vrow);
      uint4 v1 = *(const uint4*)(const void*)(vrow + 8);
      #pragma unroll
      for (int r = 0; r < 8; ++r) acc[nt][r] *= av[r];
      acc[nt] = wmma_bf16(Pa, mkfrag(v0, v1), acc[nt]);
    }
  }

  __syncthreads();
  float inv[8];
  #pragma unroll
  for (int r = 0; r < 8; ++r) {
    float l = lstat[qw][r + 8 * hi];
    inv[r] = (l > 0.f) ? (1.0f / l) : 0.f;   // fully-masked rows -> 0
  }
  // Single base pointer; per-(nt,r) offsets are immediates.
  float* obase =
      out + ((size_t)(b * SEQ) + q0 + qw * 16 + 8 * hi) * DMODEL + dw * 128 + l15;
  #pragma unroll
  for (int nt = 0; nt < 8; ++nt)
    #pragma unroll
    for (int r = 0; r < 8; ++r)
      obase[r * DMODEL + nt * 16] = acc[nt][r] * inv[r];
}

// ---------------------------------------------------------------------------
extern "C" void kernel_launch(void* const* d_in, const int* in_sizes, int n_in,
                              void* d_out, int out_size, void* d_ws, size_t ws_size,
                              hipStream_t stream) {
  (void)in_sizes; (void)n_in; (void)out_size; (void)ws_size;
  const float*         x    = (const float*)d_in[0];
  const float*         bias = (const float*)d_in[1];
  const unsigned char* kpm  = (const unsigned char*)d_in[2];  // bool mask
  const float*         Wq   = (const float*)d_in[3];
  const float*         Wk   = (const float*)d_in[4];
  const float*         Wv   = (const float*)d_in[5];
  float*               out  = (float*)d_out;

  unsigned short* Qb = (unsigned short*)d_ws;                  // [BS, D] bf16
  unsigned short* Kb = Qb + (size_t)BSROWS * DMODEL;           // [BS, D] bf16
  unsigned short* Vt = Kb + (size_t)BSROWS * DMODEL;           // [B, D, S] bf16

  proj_kernel<<<dim3(BSROWS / 128, DMODEL / 64, 3), 256, 0, stream>>>(
      x, Wq, Wk, Wv, Qb, Kb, Vt);
  attn_kernel<<<dim3(SEQ / 32, BATCH), 256, 0, stream>>>(
      Qb, Kb, Vt, bias, kpm, out);
}
